// SNN_joint_learning_module_80650895884343
// MI455X (gfx1250) — matrix operand order repaired and checked by reading, not compiled, not run on hardware
//
#include <hip/hip_runtime.h>
#include <hip/hip_bf16.h>

typedef __attribute__((ext_vector_type(16))) _Float16 v16h;
typedef __attribute__((ext_vector_type(8)))  _Float16 v8h;
typedef __attribute__((ext_vector_type(8)))  float    v8f;
typedef __attribute__((ext_vector_type(4)))  unsigned int u32x4;
typedef __attribute__((ext_vector_type(4)))  int i32x4;
typedef __attribute__((ext_vector_type(8)))  int i32x8;

#define HH   4
#define NN   512
#define DIMK 8192
#define OC   128
#define KC   64
#define NDIM (NN * DIMK)       // 4194304 elements per head
#define WHALF (OC * DIMK)      // 1048576 halfs per weight matrix
#define SCALE 0.125f           // 64^-0.5

#if __has_builtin(__builtin_amdgcn_tensor_load_to_lds) && __has_builtin(__builtin_amdgcn_s_wait_tensorcnt)
#define USE_TDM 1
#pragma message("CDNA5 probe: TDM path ENABLED (tensor_load_to_lds + s_wait_tensorcnt)")
#else
#define USE_TDM 0
#pragma message("CDNA5 probe: TDM path DISABLED (builtins absent, manual LDS staging)")
#endif
#define STR2(x) #x
#define STR(x) STR2(x)
#pragma message("CDNA5 probe: __clang_major__ = " STR(__clang_major__))

// ---------------- workspace layout (float-index offsets) ----------------
#define F_STATSACC 0u          // [2 src][4 h][sum, sumsq]           16
#define F_STATSFIN 16u         // [2 src][4 h][mean, inv_std]        16
#define F_WEFF     32u         // collapsed conv weight              512
#define F_CONST    544u        // collapsed conv bias                4
#define F_O2       548u        // conv-chain output                  512
#define F_O3       1060u       // fc output                          512
#define F_IPSUM    1572u       // [4][512]                           2048
#define F_ICSUM    3620u       // [512][64]                          32768
#define F_IPRAW    36388u      // [4][512][128] raw pic gemm         262144
#define F_CTXRAW   298532u     // [4][512][128] raw ctx gemm         262144
#define F_OUTMAT   560676u     // [512 n][512 col] attn output       262144
#define B_WF16     3291280u    // byte offset of f16 weights (16B aligned), 4 MiB

// ---------------- warp/block reduction helpers (wave32) ----------------
__device__ inline float wsum(float v) {
#pragma unroll
  for (int m = 16; m >= 1; m >>= 1) v += __shfl_xor(v, m, 32);
  return v;
}
__device__ inline float wmaxr(float v) {
#pragma unroll
  for (int m = 16; m >= 1; m >>= 1) v = fmaxf(v, __shfl_xor(v, m, 32));
  return v;
}
__device__ inline float block_sum256(float v) {   // valid result in warp 0
  __shared__ float tmp[8];
  __syncthreads();
  v = wsum(v);
  const int warp = threadIdx.x >> 5, lane = threadIdx.x & 31;
  if (!lane) tmp[warp] = v;
  __syncthreads();
  float r = 0.f;
  if (warp == 0) { r = (lane < 8) ? tmp[lane] : 0.f; r = wsum(r); }
  return r;
}

// ---------------- K0: zero stats accumulators + ic_sum ----------------
__global__ void k_init(float* ws) {
  const int i = blockIdx.x * 256 + threadIdx.x;
  if (i < 16) ws[F_STATSACC + i] = 0.f;
  if (i < NN * 64) ws[F_ICSUM + i] = 0.f;
}

// ---------------- K1: layernorm sum / sumsq per (src, h), b128 reads -------
__global__ __launch_bounds__(256) void k_stats(const float* __restrict__ pic,
                                               const float* __restrict__ ctx,
                                               float* ws) {
  const int src = blockIdx.z, h = blockIdx.y;
  const float4* p = (const float4*)((src ? ctx : pic) + (size_t)h * NDIM);
  float s = 0.f, ss = 0.f;
  for (int i = blockIdx.x * blockDim.x + threadIdx.x; i < NDIM / 4;
       i += blockDim.x * gridDim.x) {
    const float4 v = p[i];
    s += v.x + v.y + v.z + v.w;
    ss += v.x * v.x + v.y * v.y + v.z * v.z + v.w * v.w;
  }
  s = wsum(s); ss = wsum(ss);
  __shared__ float a1[8], a2[8];
  const int warp = threadIdx.x >> 5, lane = threadIdx.x & 31;
  if (!lane) { a1[warp] = s; a2[warp] = ss; }
  __syncthreads();
  if (warp == 0) {
    float x = (lane < 8) ? a1[lane] : 0.f;
    float y = (lane < 8) ? a2[lane] : 0.f;
    x = wsum(x); y = wsum(y);
    if (!lane) {
      atomicAdd(&ws[F_STATSACC + (src * HH + h) * 2 + 0], x);
      atomicAdd(&ws[F_STATSACC + (src * HH + h) * 2 + 1], y);
    }
  }
}

__global__ void k_statsfin(float* ws) {
  const int i = threadIdx.x;  // 0..7 = (src, h)
  const float M = (float)NDIM;
  float S = ws[F_STATSACC + i * 2 + 0];
  float SS = ws[F_STATSACC + i * 2 + 1];
  float mean = S / M;
  float var = SS / M - mean * mean;
  ws[F_STATSFIN + i * 2 + 0] = mean;
  ws[F_STATSFIN + i * 2 + 1] = rsqrtf(var + 1e-5f);
}

// ---------------- K2: convert W_pic / W_ctx fp32 -> f16 ----------------
__global__ void k_wconv(const float* __restrict__ wp, const float* __restrict__ wc,
                        _Float16* __restrict__ wf) {
  const int i = blockIdx.x * 256 + threadIdx.x;
  if (i < WHALF) wf[i] = (_Float16)wp[i];
  else           wf[i] = (_Float16)wc[i - WHALF];
}

// ---------------- K2b: collapse conv1(32,512) x conv2(1,32) ----------------
__global__ void k_weff(const float* __restrict__ c1w, const float* __restrict__ c1b,
                       const float* __restrict__ c2w, const float* __restrict__ c2b,
                       float* ws) {
  const int c = threadIdx.x;  // 0..511
  float s = 0.f;
  for (int o = 0; o < 32; ++o) s += c2w[o] * c1w[o * 512 + c];
  ws[F_WEFF + c] = s;
  if (c == 0) {
    float t = c2b[0];
    for (int o = 0; o < 32; ++o) t += c2w[o] * c1b[o];
    ws[F_CONST] = t;
  }
}

#if USE_TDM
// Issue a TDM load of a [OC rows x KC cols] f16 tile (row stride DIMK) into LDS.
__device__ inline void tdm_load_w(const _Float16* gsrc, unsigned lds_addr) {
  const unsigned long long ga = (unsigned long long)(size_t)gsrc;
  u32x4 g0;
  g0[0] = 1u;                                   // count=1, user mode, no gather
  g0[1] = lds_addr;                             // LDS byte address of tile start
  g0[2] = (unsigned)(ga & 0xFFFFFFFFu);         // global_addr[31:0]
  g0[3] = (unsigned)((ga >> 32) & 0x01FFFFFFu)  // global_addr[56:32]
          | (2u << 30);                         // type = 2 ("image")
  i32x8 g1;
  g1[0] = (int)(1u << 16);                      // data_size = 1 (2-byte elems)
  g1[1] = (int)((DIMK & 0xFFFF) << 16);         // tensor_dim0[15:0]
  g1[2] = (int)(((unsigned)DIMK >> 16) | ((unsigned)OC << 16)); // dim0 hi | dim1 lo
  g1[3] = (int)((unsigned)KC << 16);            // dim1 hi = 0 | tile_dim0 = KC
  g1[4] = (int)OC;                              // tile_dim1 = OC, tile_dim2 = 0
  g1[5] = (int)DIMK;                            // tensor_dim0_stride[31:0]
  g1[6] = 0;                                    // stride hi | dim1_stride lo
  g1[7] = 0;
  const i32x4 z4 = {0, 0, 0, 0};
#if __clang_major__ >= 23
  const i32x8 z8 = {0, 0, 0, 0, 0, 0, 0, 0};
  __builtin_amdgcn_tensor_load_to_lds(g0, g1, z4, z4, z8, 0);
#else
  __builtin_amdgcn_tensor_load_to_lds(g0, g1, z4, z4, 0);
#endif
}
#endif

// ---------------- K3: fused layernorm + spike + dual f16 WMMA GEMM ----------
// Grid: (N/32, H).  Block: 256 = 8 waves.
// Waves 0-3: pic GEMM, waves 4-7: ctx GEMM.
// Each wave owns a 16-row x 64-col output slab = 4 WMMA tiles.
// Weights arrive via TDM (tensor_load_to_lds), double-buffered so the fetch of
// step i+1 overlaps the spike computation + WMMA of step i.
__global__ __launch_bounds__(256) void k_gemm(const float* __restrict__ pic,
                                              const float* __restrict__ ctx,
                                              const float* __restrict__ gamma,
                                              const float* __restrict__ beta,
                                              float* ws,
                                              const _Float16* __restrict__ wf16) {
  const int tid = threadIdx.x, lane = tid & 31, wave = tid >> 5;
  const int h = blockIdx.y, row0 = blockIdx.x * 32;
  const int gsel = wave >> 2;          // 0 = pic gemm, 1 = ctx gemm
  const int w4 = wave & 3;
  const int rowt = w4 & 1;             // row tile: 16 rows
  const int colh = w4 >> 1;            // column half: 64 cols

  __shared__ alignas(32) _Float16 sA[2][32 * KC];        // spikes [src][r][k]  8 KB
  __shared__ alignas(32) _Float16 sB[2][2][OC * KC];     // [buf][gemm][o][k]  64 KB

  const float mean0 = ws[F_STATSFIN + (0 * HH + h) * 2 + 0];
  const float istd0 = ws[F_STATSFIN + (0 * HH + h) * 2 + 1];
  const float mean1 = ws[F_STATSFIN + (1 * HH + h) * 2 + 0];
  const float istd1 = ws[F_STATSFIN + (1 * HH + h) * 2 + 1];

  v8f acc[4] = {};

  const int m = lane & 15;             // M row / N col within tile
  const int hl = lane >> 4;            // lane half selector
  const int arow = rowt * 16 + m;

  // spike staging geometry: 512 float4 per source, 2 per thread
  const int r_st = (tid + 0 * 256) >> 4;          // row for q=0   (0..15)
  const int r_st2 = (tid + 1 * 256) >> 4;         // row for q=1   (16..31)
  const int k4_st = (tid & 15) * 4;               // k offset (same both q)

#if USE_TDM
  if (wave == 0) {   // prologue: fetch weights for k0 = 0 into buffer 0
    tdm_load_w(wf16 + 0 * WHALF, (unsigned)(size_t)&sB[0][0][0]);
    tdm_load_w(wf16 + 1 * WHALF, (unsigned)(size_t)&sB[0][1][0]);
  }
#endif

  for (int it = 0; it < DIMK / KC; ++it) {
    const int k0 = it * KC;
    const int buf = it & 1;
    __syncthreads();   // prior step done reading sA and sB[buf^1]
    // ---- stage spikes for both sources: 32 rows x 64 k, b128 reads ----
#pragma unroll
    for (int src = 0; src < 2; ++src) {
      const float* sp = src ? ctx : pic;
      const float mean = src ? mean1 : mean0;
      const float istd = src ? istd1 : istd0;
#pragma unroll
      for (int q = 0; q < 2; ++q) {
        const int r = q ? r_st2 : r_st;
        const size_t gi = (size_t)(h * NN + row0 + r) * DIMK + k0 + k4_st;
        const float4 v = *(const float4*)(sp + gi);
        const float4 ga = *(const float4*)(gamma + k0 + k4_st);
        const float4 be = *(const float4*)(beta + k0 + k4_st);
        union { _Float16 hx[4]; uint2 u; } pk;
        pk.hx[0] = (_Float16)(((v.x - mean) * istd * ga.x + be.x) > 1.0f ? 1.0f : 0.0f);
        pk.hx[1] = (_Float16)(((v.y - mean) * istd * ga.y + be.y) > 1.0f ? 1.0f : 0.0f);
        pk.hx[2] = (_Float16)(((v.z - mean) * istd * ga.z + be.z) > 1.0f ? 1.0f : 0.0f);
        pk.hx[3] = (_Float16)(((v.w - mean) * istd * ga.w + be.w) > 1.0f ? 1.0f : 0.0f);
        *(uint2*)&sA[src][r * KC + k4_st] = pk.u;
      }
    }
#if USE_TDM
    if (wave == 0) {
      if (k0 + KC < DIMK) {
        // prefetch next chunk into the other buffer, then wait for the pair
        // belonging to the *current* buffer (allow 2 newly issued outstanding)
        tdm_load_w(wf16 + 0 * WHALF + (k0 + KC), (unsigned)(size_t)&sB[buf ^ 1][0][0]);
        tdm_load_w(wf16 + 1 * WHALF + (k0 + KC), (unsigned)(size_t)&sB[buf ^ 1][1][0]);
        __builtin_amdgcn_s_wait_tensorcnt(2);
      } else {
        __builtin_amdgcn_s_wait_tensorcnt(0);
      }
    }
#else
    // ---- fallback: stage f16 weight chunks manually, 16B vectorized ----
#pragma unroll
    for (int g = 0; g < 2; ++g) {
#pragma unroll
      for (int q = 0; q < 4; ++q) {
        const int cc = tid + q * 256;            // 0..1023 16-byte chunks
        const int o = cc >> 3, kk = (cc & 7) * 8;
        const uint4 wv =
            *(const uint4*)&wf16[(size_t)g * WHALF + (size_t)o * DIMK + k0 + kk];
        *(uint4*)&sB[buf][g][o * KC + kk] = wv;
      }
    }
#endif
    if (k0 + KC < DIMK)   // pull next activation chunk toward the WGP caches
      __builtin_prefetch(&pic[(size_t)(h * NN + row0 + (tid >> 3)) * DIMK + k0 + KC], 0, 3);
    __syncthreads();   // sA staged + sB[buf] landed (TENSORcnt waited by wave 0)

    // ---- WMMA: 2 K-substeps of 32, 4 column tiles each ----
#pragma unroll
    for (int ks = 0; ks < KC; ks += 32) {
      // A fragment (16x32 f16), contiguous k order, same as B:
      //   a[0..7]  = A[m][ks + 16*hl + 0..7]
      //   a[8..15] = A[m][ks + 16*hl + 8..15]
      union { v16h v; v8h h2[2]; } ua;
      ua.h2[0] = *(const v8h*)&sA[gsel][arow * KC + ks + 16 * hl];
      ua.h2[1] = *(const v8h*)&sA[gsel][arow * KC + ks + 16 * hl + 8];
#pragma unroll
      for (int ct = 0; ct < 4; ++ct) {
        // B fragment (32x16 f16): b[0..15] = W[o = colbase+m][ks + 16*hl + 0..15]
        const int orow = colh * 64 + ct * 16 + m;
        const v16h bf = *(const v16h*)&sB[buf][gsel][orow * KC + ks + 16 * hl];
        acc[ct] = __builtin_amdgcn_wmma_f32_16x16x32_f16(
            false, ua.v, false, bf, (short)0, acc[ct], false, false);
      }
    }
  }

  // ---- store raw accumulators (D layout: VGPR v -> M = v + 8*hl, N = m) ----
  float* dst = ws + (gsel ? F_CTXRAW : F_IPRAW);
#pragma unroll
  for (int ct = 0; ct < 4; ++ct) {
#pragma unroll
    for (int v = 0; v < 8; ++v) {
      const int row = row0 + rowt * 16 + v + 8 * hl;
      const int col = colh * 64 + ct * 16 + m;
      dst[(size_t)(h * NN + row) * OC + col] = acc[ct][v];
    }
  }
}

// ---------------- K4: per-(h,n) bias/scale, softmax-attention, sums --------
__global__ __launch_bounds__(256) void k_row(const float* __restrict__ b_pic,
                                             const float* __restrict__ b_ctx,
                                             float* ws) {
  const int tid = threadIdx.x, warp = tid >> 5, lane = tid & 31;
  const int row = blockIdx.x * 8 + warp;   // 0..2047 = h*512 + n
  const int h = row >> 9, n = row & 511;
  const float* ipr = ws + F_IPRAW + (size_t)row * OC;
  const float* cxr = ws + F_CTXRAW + (size_t)row * OC;
  float ipv[4], cv[4];
#pragma unroll
  for (int q = 0; q < 4; ++q) {
    const int o = lane + q * 32;
    ipv[q] = (ipr[o] + b_pic[o]) * SCALE;
    cv[q] = cxr[o] + b_ctx[o];
  }
  const float s_ic = wsum(cv[0] + cv[1]);                  // sum of ic (o<64)
  const float s_V = wsum(cv[2] + cv[3]);                   // sum of V  (o>=64)
  const float ipt = wsum(ipv[0] + ipv[1] + ipv[2] + ipv[3]);
  float sim[4], mx = -3.0e38f;
#pragma unroll
  for (int q = 0; q < 4; ++q) { sim[q] = ipv[q] * s_ic; mx = fmaxf(mx, sim[q]); }
  mx = wmaxr(mx);
  float e[4], es = 0.f;
#pragma unroll
  for (int q = 0; q < 4; ++q) { e[q] = __expf(sim[q] - mx); es += e[q]; }
  es = wsum(es);
  const float inv = 1.0f / es;
#pragma unroll
  for (int q = 0; q < 4; ++q) {
    const int o = lane + q * 32;
    ws[F_OUTMAT + (size_t)n * 512 + h * OC + o] = e[q] * inv * s_V;
  }
  if (!lane) ws[F_IPSUM + h * NN + n] = ipt;
  atomicAdd(&ws[F_ICSUM + n * 64 + lane], cv[0]);          // ic_sum over h
  atomicAdd(&ws[F_ICSUM + n * 64 + lane + 32], cv[1]);
}

// ---------------- K5: collapsed conv chain: o2[col] = const + w_eff . out[:,col]
__global__ __launch_bounds__(256) void k_o2(float* ws) {
  const int col = blockIdx.x;
  float s = 0.f;
  for (int n = threadIdx.x; n < 512; n += 256)
    s += ws[F_WEFF + n] * ws[F_OUTMAT + (size_t)n * 512 + col];
  s = block_sum256(s);
  if (threadIdx.x == 0) ws[F_O2 + col] = ws[F_CONST] + s;
}

// ---------------- K6: fc matvec: o3[j] = fc_b[j] + o2 . fc_w[j,:] ----------
__global__ __launch_bounds__(256) void k_o3(float* ws, const float* __restrict__ fcw,
                                            const float* __restrict__ fcb) {
  const int j = blockIdx.x;
  float s = 0.f;
  for (int l = threadIdx.x; l < 512; l += 256)
    s += ws[F_O2 + l] * fcw[(size_t)j * 512 + l];
  s = block_sum256(s);
  if (threadIdx.x == 0) ws[F_O3 + j] = s + fcb[j];
}

// ---------------- K7: broadcast add, 4->64 linear interp, ReLUs ------------
__global__ void k_final(const float* ws, float* __restrict__ out) {
  const int n = blockIdx.x, t = threadIdx.x;   // t in 0..63
  const float src = fmaxf((t + 0.5f) * (4.0f / 64.0f) - 0.5f, 0.0f);
  int i0 = (int)floorf(src); i0 = i0 > 3 ? 3 : i0;
  const int i1 = (i0 + 1 > 3) ? 3 : i0 + 1;
  const float w = src - (float)i0;
  const float o3 = ws[F_O3 + n];
  const float v0 = o3 + ws[F_IPSUM + i0 * NN + n];
  const float v1 = o3 + ws[F_IPSUM + i1 * NN + n];
  const float itp = v0 * (1.0f - w) + v1 * w;
  const float r = fmaxf(itp, 0.0f) + ws[F_ICSUM + n * 64 + t];
  out[n * 64 + t] = fmaxf(r, 0.0f);
}

// ---------------- launch ----------------
extern "C" void kernel_launch(void* const* d_in, const int* in_sizes, int n_in,
                              void* d_out, int out_size, void* d_ws, size_t ws_size,
                              hipStream_t stream) {
  const float* pic   = (const float*)d_in[0];
  const float* ctx   = (const float*)d_in[1];
  const float* gamma = (const float*)d_in[2];
  const float* beta  = (const float*)d_in[3];
  const float* Wpic  = (const float*)d_in[4];
  const float* b_pic = (const float*)d_in[5];
  const float* Wctx  = (const float*)d_in[6];
  const float* b_ctx = (const float*)d_in[7];
  const float* c1w   = (const float*)d_in[8];
  const float* c1b   = (const float*)d_in[9];
  const float* c2w   = (const float*)d_in[10];
  const float* c2b   = (const float*)d_in[11];
  const float* fcw   = (const float*)d_in[12];
  const float* fcb   = (const float*)d_in[13];
  float* ws = (float*)d_ws;
  _Float16* wf16 = (_Float16*)((char*)d_ws + B_WF16);
  float* out = (float*)d_out;

  k_init<<<128, 256, 0, stream>>>(ws);
  k_stats<<<dim3(128, HH, 2), 256, 0, stream>>>(pic, ctx, ws);
  k_statsfin<<<1, 8, 0, stream>>>(ws);
  k_wconv<<<(2 * WHALF) / 256, 256, 0, stream>>>(Wpic, Wctx, wf16);
  k_weff<<<1, 512, 0, stream>>>(c1w, c1b, c2w, c2b, ws);
  k_gemm<<<dim3(NN / 32, HH), 256, 0, stream>>>(pic, ctx, gamma, beta, ws, wf16);
  k_row<<<256, 256, 0, stream>>>(b_pic, b_ctx, ws);
  k_o2<<<512, 256, 0, stream>>>(ws);
  k_o3<<<512, 256, 0, stream>>>(ws, fcw, fcb);
  k_final<<<512, 64, 0, stream>>>(ws, out);
}
